// CTRGC_1228360647383
// MI455X (gfx1250) — compile-verified, matrix operand order checked by reading, not run on hardware
//
#include <hip/hip_runtime.h>
#include <math.h>

typedef __attribute__((ext_vector_type(16))) __bf16 v16bf;
typedef __attribute__((ext_vector_type(8)))  float  v8f;
typedef __attribute__((ext_vector_type(4)))  float  v4f;
typedef __attribute__((ext_vector_type(4)))  unsigned int v4u;
typedef __attribute__((ext_vector_type(2)))  unsigned int v2u;

#define N_  256
#define C_  64
#define T_  64
#define V_  25
#define R_  8
#define O_  64
#define TC_ 16               // t-chunk size
#define NCHUNK (T_/TC_)      // 4 chunks
#define TV_ (TC_*V_)         // 400 columns per chunk
#define NT_ 512              // threads per block (16 waves)

// LDS layout: bf16 (ushort) arrays first; all fragment bases are 16B aligned
// so operand quarters load as ds_load_b128.
struct SharedMem {
  unsigned short M [O_*32*32];   // [o][u][v]  bf16, u,v padded 25->32 (zeros)   128 KB
  unsigned short x3[O_*TC_*32];  // [o][tl][v] bf16, v padded (zeros)             64 KB
  unsigned short xb[TC_*V_*C_];  // [tl*25+v][c] bf16 transposed x chunk          50 KB
  float xm [C_*V_];              // mean over T
  float x1 [R_*V_];
  float x2 [R_*V_];
  float aff[R_*V_*V_];
  float w1 [R_*C_];
  float w2 [R_*C_];
  float w4 [O_*R_];
  float b1 [R_];
  float b2 [R_];
  float b3 [O_];
  float b4 [O_];
  float Am [V_*V_];
  float alpha;
};

__device__ __forceinline__ unsigned short bfbits(float f) {
  __bf16 h = (__bf16)f;
  return __builtin_bit_cast(unsigned short, h);
}
__device__ __forceinline__ unsigned packbf(float a, float b) {
  return (unsigned)bfbits(a) | ((unsigned)bfbits(b) << 16);
}

// Fragment: VGPRs 0..3 = K(8kh .. 8kh+7), VGPRs 4..7 = K(16+8kh .. 23+8kh)
// -> two contiguous 16B quarters -> two ds_load_b128.
union Frag { v16bf v; v4u q[2]; unsigned u[8]; };

__global__ __launch_bounds__(NT_) void ctrgc_fused(
    const float* __restrict__ x,  const float* __restrict__ Am,
    const float* __restrict__ alpha,
    const float* __restrict__ w1, const float* __restrict__ b1,
    const float* __restrict__ w2, const float* __restrict__ b2,
    const float* __restrict__ w3, const float* __restrict__ b3,
    const float* __restrict__ w4, const float* __restrict__ b4,
    float* __restrict__ out)
{
  extern __shared__ char smem_raw[];
  SharedMem& s = *reinterpret_cast<SharedMem*>(smem_raw);

  const int tid  = threadIdx.x;
  const int n    = blockIdx.x;
  const int lane = tid & 31;
  const int wave = tid >> 5;    // 0..15
  const int ln   = lane & 15;   // row/col index within 16
  const int kh   = lane >> 4;   // K-half select

  // ---- Phase 0: zero padded bf16 buffers, stage constants to LDS ----
  {
    unsigned* M32  = (unsigned*)s.M;
    unsigned* x332 = (unsigned*)s.x3;
    for (int i = tid; i < O_*32*32/2;  i += NT_) M32[i]  = 0u;
    for (int i = tid; i < O_*TC_*32/2; i += NT_) x332[i] = 0u;
    for (int i = tid; i < R_*C_; i += NT_) { s.w1[i] = w1[i]; s.w2[i] = w2[i]; s.w4[i] = w4[i]; }
    for (int i = tid; i < V_*V_; i += NT_) s.Am[i] = Am[i];
    if (tid < O_) { s.b3[tid] = b3[tid]; s.b4[tid] = b4[tid]; }
    if (tid < R_) { s.b1[tid] = b1[tid]; s.b2[tid] = b2[tid]; }
    if (tid == 0) s.alpha = alpha[0];
  }

  // ---- Phase 0b: w3 A-fragments for this wave's o-tile (float4 loads) ----
  const int myot = wave & 3;            // each wave owns one 16-row o-tile for GEMM1
  Frag wa[2];                           // 2 K-steps over C=64
  {
    const v4f* wrow = (const v4f*)(w3 + (myot*16 + ln)*C_);
    #pragma unroll
    for (int st = 0; st < 2; ++st) {
      int b = st*8 + 2*kh;              // float4 index of K(8kh) within this K-step
      v4f f0 = wrow[b],     f1 = wrow[b + 1];
      v4f f2 = wrow[b + 4], f3 = wrow[b + 5];
      wa[st].u[0] = packbf(f0.x, f0.y);
      wa[st].u[1] = packbf(f0.z, f0.w);
      wa[st].u[2] = packbf(f1.x, f1.y);
      wa[st].u[3] = packbf(f1.z, f1.w);
      wa[st].u[4] = packbf(f2.x, f2.y);
      wa[st].u[5] = packbf(f2.z, f2.w);
      wa[st].u[6] = packbf(f3.x, f3.y);
      wa[st].u[7] = packbf(f3.z, f3.w);
    }
  }

  // ---- Phase 1: xm[c][v] = mean over T (first streaming pass over x[n]) ----
  for (int p = tid; p < C_*V_; p += NT_) {
    int c = p / V_, v = p % V_;
    const float* px = x + (((size_t)n*C_ + c)*T_)*V_ + v;
    float acc = 0.f;
    #pragma unroll 8
    for (int t = 0; t < T_; ++t) acc += px[t*V_];
    s.xm[p] = acc * (1.0f/T_);
  }
  __syncthreads();

  // ---- Phase 2a: x1 = w1*xm + b1 ; x2 = w2*xm + b2 ----
  for (int q = tid; q < 2*R_*V_; q += NT_) {
    int sel = (q >= R_*V_) ? 1 : 0;
    int j = q - sel*R_*V_;
    int r = j / V_, v = j % V_;
    const float* wrow = (sel ? s.w2 : s.w1) + r*C_;
    float acc = sel ? s.b2[r] : s.b1[r];
    #pragma unroll 8
    for (int c = 0; c < C_; ++c) acc += wrow[c] * s.xm[c*V_ + v];
    (sel ? s.x2 : s.x1)[j] = acc;
  }
  __syncthreads();

  // ---- Phase 2b: aff[r,u,v] = tanh(x1[r,u] - x2[r,v]) ----
  for (int q = tid; q < R_*V_*V_; q += NT_) {
    int r = q / (V_*V_), rem = q % (V_*V_);
    int u = rem / V_, v = rem % V_;
    s.aff[q] = tanhf(s.x1[r*V_ + u] - s.x2[r*V_ + v]);
  }
  __syncthreads();

  // ---- Phase 2c: M[o,u,v] = (w4·aff + b4)*alpha + A   (rem outer, o inner) ----
  {
    float alp = s.alpha;
    for (int rem = tid; rem < V_*V_; rem += NT_) {
      int u = rem / V_, v = rem % V_;
      float av[R_];
      #pragma unroll
      for (int r = 0; r < R_; ++r) av[r] = s.aff[r*V_*V_ + rem];
      float Aval = s.Am[rem];
      unsigned short* mp = &s.M[u*32 + v];
      for (int o = 0; o < O_; ++o) {
        float acc = s.b4[o];
        #pragma unroll
        for (int r = 0; r < R_; ++r) acc += s.w4[o*R_ + r] * av[r];
        mp[o*32*32] = bfbits(acc * alp + Aval);
      }
    }
  }
  __syncthreads();

  const v4u* xbq = (const v4u*)s.xb;
  const v4u* x3q = (const v4u*)s.x3;
  const v4u* Mq  = (const v4u*)s.M;

  // ---- Preload this wave's GEMM2 M-fragments (invariant across chunks) ----
  Frag mfr[4][2];                        // o = 16*i + wave, u-tiles 0/1
  #pragma unroll
  for (int i = 0; i < 4; ++i) {
    int o = i*16 + wave;
    #pragma unroll
    for (int ut = 0; ut < 2; ++ut) {
      int qb = ((o*32 + ut*16 + ln)*32 + 8*kh) >> 3;
      mfr[i][ut].q[0] = Mq[qb];
      mfr[i][ut].q[1] = Mq[qb + 2];
    }
  }

  // ---- Phase 3: chunked over T (second streaming pass over x[n]) ----
  for (int tc = 0; tc < NCHUNK; ++tc) {
    __syncthreads();   // xb rewrite vs previous chunk's GEMM1 reads

    // 3a: load 4 c-rows of the chunk as float4 (coalesced), pack, ds_store_b64
    for (int q = tid; q < (C_/4)*(TV_/4); q += NT_) {
      int cg = q / (TV_/4), i4 = q % (TV_/4);   // c = 4*cg
      const float* base = x + (((size_t)n*C_ + 4*cg)*T_ + tc*TC_)*V_;
      v4f g0 = *(const v4f*)(base + 0*T_*V_ + 4*i4);
      v4f g1 = *(const v4f*)(base + 1*T_*V_ + 4*i4);
      v4f g2 = *(const v4f*)(base + 2*T_*V_ + 4*i4);
      v4f g3 = *(const v4f*)(base + 3*T_*V_ + 4*i4);
      int rr = i4*4;                            // flattened tl*25+v
      #pragma unroll
      for (int k = 0; k < 4; ++k) {
        v2u pk;
        pk.x = packbf(g0[k], g1[k]);
        pk.y = packbf(g2[k], g3[k]);
        *(v2u*)(s.xb + (rr + k)*C_ + 4*cg) = pk;   // ds_store_b64
      }
    }
    // hint-prefetch next chunk (global_prefetch_b8)
    if (tc + 1 < NCHUNK) {
      for (int i = tid; i < C_; i += NT_) {
        const float* pp = x + (((size_t)n*C_ + i)*T_ + (tc+1)*TC_)*V_;
        __builtin_prefetch(pp,       0, 0);
        __builtin_prefetch(pp + 100, 0, 0);
        __builtin_prefetch(pp + 200, 0, 0);
        __builtin_prefetch(pp + 300, 0, 0);
      }
    }
    __syncthreads();

    // 3b: GEMM1  x3[o, col] = w3[o,c] * xb[c, col] + b3   (K = 64 = 2 WMMA steps)
    for (int ct = (wave >> 2); ct < 25; ct += 4) {   // this wave's columns, ot fixed
      int colg = ct*16 + ln;                          // column 0..399 (tl*25+v)
      Frag bf0, bf1;
      int qi0 = (colg*C_ +  0 + 8*kh) >> 3;           // v4u index, K-step 0
      int qi1 = (colg*C_ + 32 + 8*kh) >> 3;           // K-step 1
      bf0.q[0] = xbq[qi0];  bf0.q[1] = xbq[qi0 + 2];  // ds_load_b128 x2
      bf1.q[0] = xbq[qi1];  bf1.q[1] = xbq[qi1 + 2];
      v8f acc = {};
      acc = __builtin_amdgcn_wmma_f32_16x16x32_bf16(false, wa[0].v, false, bf0.v,
                                                    (short)0, acc, false, false);
      acc = __builtin_amdgcn_wmma_f32_16x16x32_bf16(false, wa[1].v, false, bf1.v,
                                                    (short)0, acc, false, false);
      int tl = colg / V_, v = colg % V_;
      #pragma unroll
      for (int j = 0; j < 8; ++j) {
        int o = myot*16 + j + 8*kh;                   // C/D row for VGPR j
        s.x3[(o*TC_ + tl)*32 + v] = bfbits(acc[j] + s.b3[o]);
      }
    }
    __syncthreads();

    // 3c: GEMM2  out[o, t, u] = sum_v x3[o,t,v] * M[o,u,v]   (K = 25 padded to 32)
    #pragma unroll
    for (int i = 0; i < 4; ++i) {
      int o = i*16 + wave;
      Frag af;
      int qa = ((o*TC_ + ln)*32 + 8*kh) >> 3;
      af.q[0] = x3q[qa];  af.q[1] = x3q[qa + 2];      // ds_load_b128 x2
      #pragma unroll
      for (int ut = 0; ut < 2; ++ut) {
        v8f acc = {};
        acc = __builtin_amdgcn_wmma_f32_16x16x32_bf16(false, af.v, false, mfr[i][ut].v,
                                                      (short)0, acc, false, false);
        int u = ut*16 + ln;
        if (u < V_) {
          #pragma unroll
          for (int j = 0; j < 8; ++j) {
            int t = tc*TC_ + j + 8*kh;
            out[(((size_t)n*O_ + o)*T_ + t)*V_ + u] = acc[j];
          }
        }
      }
    }
  }
}

extern "C" void kernel_launch(void* const* d_in, const int* in_sizes, int n_in,
                              void* d_out, int out_size, void* d_ws, size_t ws_size,
                              hipStream_t stream) {
  (void)in_sizes; (void)n_in; (void)d_ws; (void)ws_size; (void)out_size;
  const float* x  = (const float*)d_in[0];
  const float* A  = (const float*)d_in[1];
  const float* al = (const float*)d_in[2];
  const float* w1 = (const float*)d_in[3];
  const float* b1 = (const float*)d_in[4];
  const float* w2 = (const float*)d_in[5];
  const float* b2 = (const float*)d_in[6];
  const float* w3 = (const float*)d_in[7];
  const float* b3 = (const float*)d_in[8];
  const float* w4 = (const float*)d_in[9];
  const float* b4 = (const float*)d_in[10];
  float* out = (float*)d_out;

  const size_t shmem = sizeof(SharedMem);   // ~279 KB < 320 KB per WGP
  (void)hipFuncSetAttribute(reinterpret_cast<const void*>(ctrgc_fused),
                            hipFuncAttributeMaxDynamicSharedMemorySize, (int)shmem);
  ctrgc_fused<<<N_, NT_, shmem, stream>>>(x, A, al, w1, b1, w2, b2, w3, b3, w4, b4, out);
}
